// GVP_encoder_103079215118
// MI455X (gfx1250) — compile-verified, matrix operand order checked by reading, not run on hardware
//
#include <hip/hip_runtime.h>
#include <math.h>

// ---------------------------------------------------------------------------
// Types for WMMA (gfx1250, wave32)
// ---------------------------------------------------------------------------
typedef __bf16 bf16_t;
typedef __attribute__((ext_vector_type(16))) __bf16 v16bf;
typedef __attribute__((ext_vector_type(8)))  float  v8f;

#define EPSF  1e-8f

// K pattern for 16-bit WMMA operands (A 16x32 / B 32x16), per ISA 7.12.2:
//   lane half 0 covers K {0..7, 16..23}, half 1 covers K {8..15, 24..31};
//   vector elem j: r=j>>1, e=j&1; r<4 -> K=r*2+e, r>=4 -> K=16+(r-4)*2+e; +8*half
__device__ __host__ inline int wmma_kpat(int half, int r, int e) {
    return ((r < 4) ? (r * 2 + e) : (16 + (r - 4) * 2 + e)) + 8 * half;
}
// Inverse: given k within a 32-chunk and column index i (0..15), slot in a
// 512-element fragment chunk laid out [lane][elem].
__device__ inline int wmma_slot(int k, int i) {
    int kc = k >> 5, kk = k & 31;
    int half = (kk >> 3) & 1;
    int b = kk - 8 * half;                       // {0..7} U {16..23}
    int r = (b < 8) ? (b >> 1) : (4 + ((b - 16) >> 1));
    int e = b & 1;
    return kc * 512 + (half * 16 + i) * 16 + (r * 2 + e);
}

// ---------------------------------------------------------------------------
// Generic helper kernels
// ---------------------------------------------------------------------------
__global__ void fzero_kernel(float* p, long n) {
    long i = (long)blockIdx.x * blockDim.x + threadIdx.x;
    if (i < n) p[i] = 0.0f;
}

__global__ void cnt_scatter_kernel(const int* eidx, float* cnt, int E) {
    int e = blockIdx.x * blockDim.x + threadIdx.x;
    if (e < E) atomicAdd(&cnt[eidx[E + e]], 1.0f);
}

// Pack a [Mreal x Kreal] fp32 weight matrix into bf16 WMMA A-fragments:
// out[tile][kc][lane][elem], tile over M (16 rows), kc over K (32 cols).
__global__ void pack_wmma_a_kernel(const float* W, bf16_t* out,
                                   int Mreal, int Kreal, int ntile, int nkc) {
    int idx = blockIdx.x * blockDim.x + threadIdx.x;
    int total = ntile * nkc * 512;
    if (idx >= total) return;
    int elem = idx & 15;
    int lane = (idx >> 4) & 31;
    int kc   = (idx >> 9) % nkc;
    int tile = (idx >> 9) / nkc;
    int M = tile * 16 + (lane & 15);             // A: M = lane&15 for both halves
    int half = lane >> 4;
    int K = kc * 32 + wmma_kpat(half, elem >> 1, elem & 1);
    float v = (M < Mreal && K < Kreal) ? W[M * Kreal + K] : 0.0f;
    out[idx] = (bf16_t)v;
}

// Residual add: x += d * (cnt ? 1/max(cnt,1) : 1)
__global__ void resadd_kernel(float* x, const float* d, const float* cnt,
                              long n, int dim) {
    long i = (long)blockIdx.x * blockDim.x + threadIdx.x;
    if (i >= n) return;
    float s = 1.0f;
    if (cnt) s = 1.0f / fmaxf(cnt[i / dim], 1.0f);
    x[i] += d[i] * s;
}

// ---------------------------------------------------------------------------
// LayerNorm tuple: s -> LN(s)*w+b ; v -> v / sqrt(mean_vd(max(|v_d|^2, eps)))
// One block per item. in may alias out.
// ---------------------------------------------------------------------------
__global__ void ln_kernel(const float* in_s, const float* in_v,
                          float* out_s, float* out_v,
                          const float* w, const float* b,
                          int sdim, int vdim) {
    int i = blockIdx.x;
    int t = threadIdx.x;
    __shared__ float ss[128];
    __shared__ float sv[64];
    __shared__ float red[3];
    if (t < sdim) ss[t] = in_s[(long)i * sdim + t];
    if (t < vdim * 3) sv[t] = in_v[(long)i * vdim * 3 + t];
    __syncthreads();
    if (t == 0) {
        float mu = 0.f;
        for (int c = 0; c < sdim; ++c) mu += ss[c];
        mu /= (float)sdim;
        float var = 0.f;
        for (int c = 0; c < sdim; ++c) { float d = ss[c] - mu; var += d * d; }
        var /= (float)sdim;
        red[0] = mu;
        red[1] = rsqrtf(var + 1e-5f);
        float vs = 0.f;
        for (int h = 0; h < vdim; ++h) {
            float q = sv[h*3]*sv[h*3] + sv[h*3+1]*sv[h*3+1] + sv[h*3+2]*sv[h*3+2];
            vs += fmaxf(q, EPSF);
        }
        red[2] = rsqrtf(vs / (float)vdim);
    }
    __syncthreads();
    if (t < sdim) out_s[(long)i * sdim + t] = (ss[t] - red[0]) * red[1] * w[t] + b[t];
    if (t < vdim * 3) out_v[(long)i * vdim * 3 + t] = sv[t] * red[2];
}

// ---------------------------------------------------------------------------
// Generic GVP (block per item, scalar VALU; used for the small GEMMs)
//   vh = wh @ v ; vn = sqrt(max(|vh|^2,eps))
//   s_pre = ws_w @ [s, vn] + ws_b
//   v_out = (wv @ vh) * sigmoid(wsv_w @ s_pre + wsv_b)   (if vo > 0)
//   s_out = relu? relu(s_pre) : s_pre
// ---------------------------------------------------------------------------
__global__ void gvp_kernel(const float* in_s, const float* in_v,
                           float* out_s, float* out_v,
                           const float* wh, const float* ws_w, const float* ws_b,
                           const float* wv, const float* wsv_w, const float* wsv_b,
                           int si, int vi, int so, int vo, int h, int relu) {
    int i = blockIdx.x;
    int t = threadIdx.x;
    const int bd = blockDim.x;
    __shared__ float sin[400];
    __shared__ float vin[128];
    __shared__ float vhs[128];
    __shared__ float vns[64];
    __shared__ float spre[400];
    for (int c = t; c < si; c += bd) sin[c] = in_s[(long)i * si + c];
    for (int c = t; c < vi * 3; c += bd) vin[c] = in_v[(long)i * vi * 3 + c];
    __syncthreads();
    for (int q = t; q < h * 3; q += bd) {
        int hh = q / 3, d = q % 3;
        float a = 0.f;
        for (int c = 0; c < vi; ++c) a += wh[hh * vi + c] * vin[c * 3 + d];
        vhs[q] = a;
    }
    __syncthreads();
    for (int q = t; q < h; q += bd) {
        float a0 = vhs[q*3], a1 = vhs[q*3+1], a2 = vhs[q*3+2];
        vns[q] = sqrtf(fmaxf(a0*a0 + a1*a1 + a2*a2, EPSF));
    }
    __syncthreads();
    int kin = si + h;
    for (int o = t; o < so; o += bd) {
        float a = ws_b[o];
        const float* wr = ws_w + (long)o * kin;
        for (int c = 0; c < si; ++c) a += wr[c] * sin[c];
        for (int c = 0; c < h; ++c) a += wr[si + c] * vns[c];
        spre[o] = a;
    }
    __syncthreads();
    if (vo > 0) {
        for (int o = t; o < vo; o += bd) {
            float g = wsv_b[o];
            const float* gr = wsv_w + (long)o * so;
            for (int c = 0; c < so; ++c) g += gr[c] * spre[c];
            float sig = 1.0f / (1.0f + expf(-g));
            for (int d = 0; d < 3; ++d) {
                float a = 0.f;
                for (int c = 0; c < h; ++c) a += wv[o * h + c] * vhs[c * 3 + d];
                out_v[(long)i * vo * 3 + o * 3 + d] = sig * a;
            }
        }
    }
    for (int o = t; o < so; o += bd)
        out_s[(long)i * so + o] = relu ? fmaxf(spre[o], 0.f) : spre[o];
}

// ---------------------------------------------------------------------------
// Fused edge-message kernel (c0 -> relu -> c1 -> relu -> c2) with WMMA.
// One wave32 per 16-edge tile. Scalar GEMMs run on v_wmma_f32_16x16x32_bf16.
// Results scatter-added into dhs/dhv (segment sum; divided by cnt later).
// soL has 112 padded columns, so the accumulator writeback is unconditional
// (M <= 111); the bias index is clamped to stay in-bounds and pad rows are
// never read back (consumers use k < 100).
// ---------------------------------------------------------------------------
__global__ __launch_bounds__(32)
void edge_msg_kernel(const int* eidx, int E,
                     const float* s, const float* v,          // [N*100],[N*48]
                     const float* es, const float* ev,        // [E*32],[E*3]
                     const bf16_t* pA0, const bf16_t* pA1, const bf16_t* pA2,
                     const float* wh0, const float* b0, const float* wsvw0,
                     const float* wsvb0, const float* wv0,
                     const float* wh1, const float* b1, const float* wsvw1,
                     const float* wsvb1, const float* wv1,
                     const float* wh2, const float* b2, const float* wsvw2,
                     const float* wsvb2, const float* wv2,
                     float* dhs, float* dhv) {
    const int lane = threadIdx.x;
    const int half = lane >> 4;
    const int base = blockIdx.x * 16;

    __shared__ __align__(32) bf16_t Xf[9 * 512];   // B fragments (bf16)
    __shared__ float mvL[16][33][3];
    __shared__ float vhL[16][33][3];
    __shared__ float soL[16][112];
    __shared__ float voL[16][16][3];
    __shared__ int eL[16], srcL[16], dstL[16];

    if (lane < 16) {
        int e = base + lane; if (e >= E) e = E - 1;
        eL[lane] = e;
        srcL[lane] = eidx[e];
        dstL[lane] = eidx[E + e];
    }
    __syncthreads();

    // ---- stage 0 inputs: X = [s_src(100) | es(32) | s_dst(100) | vn(33)] K=265 pad 288
    for (int t = lane; t < 9 * 512; t += 32) Xf[t] = (bf16_t)0.f;
    __syncthreads();
    for (int t = lane; t < 232 * 16; t += 32) {
        int k = t >> 4, i = t & 15;
        float val;
        if (k < 100)      val = s[(long)srcL[i] * 100 + k];
        else if (k < 132) val = es[(long)eL[i] * 32 + (k - 100)];
        else              val = s[(long)dstL[i] * 100 + (k - 132)];
        Xf[wmma_slot(k, i)] = (bf16_t)val;
    }
    for (int t = lane; t < 33 * 3 * 16; t += 32) {
        int i = t & 15, rest = t >> 4;
        int h = rest / 3, d = rest % 3;
        float val;
        if (h < 16)       val = v[(long)srcL[i] * 48 + h * 3 + d];
        else if (h == 16) val = ev[(long)eL[i] * 3 + d];
        else              val = v[(long)dstL[i] * 48 + (h - 17) * 3 + d];
        mvL[i][h][d] = val;
    }
    __syncthreads();

    // vh0 = wh0(33x33) @ mv ; vn -> K = 232+h
    for (int t = lane; t < 33 * 16; t += 32) {
        int i = t & 15, h = t >> 4;
        float a0 = 0.f, a1 = 0.f, a2 = 0.f;
        for (int c = 0; c < 33; ++c) {
            float w = wh0[h * 33 + c];
            a0 += w * mvL[i][c][0]; a1 += w * mvL[i][c][1]; a2 += w * mvL[i][c][2];
        }
        vhL[i][h][0] = a0; vhL[i][h][1] = a1; vhL[i][h][2] = a2;
        Xf[wmma_slot(232 + h, i)] = (bf16_t)sqrtf(fmaxf(a0*a0 + a1*a1 + a2*a2, EPSF));
    }
    __syncthreads();

    // GEMM c0: 7 out-tiles x 9 K-chunks of WMMA bf16
    for (int tile = 0; tile < 7; ++tile) {
        v8f acc = {};
        for (int kc = 0; kc < 9; ++kc) {
            v16bf a = ((const v16bf*)pA0)[(tile * 9 + kc) * 32 + lane];
            v16bf bb = *(const v16bf*)&Xf[kc * 512 + lane * 16];
            acc = __builtin_amdgcn_wmma_f32_16x16x32_bf16(
                false, a, false, bb, (short)0, acc, false, false);
        }
        int i = lane & 15;
        #pragma unroll
        for (int r = 0; r < 8; ++r) {
            int M = tile * 16 + r + 8 * half;          // <= 111, in-bounds of soL
            soL[i][M] = acc[r] + b0[M < 100 ? M : 99]; // pad rows never read back
        }
    }
    __syncthreads();

    // gate + v_out for c0 (gate uses pre-relu scalar)
    for (int t = lane; t < 16 * 16; t += 32) {
        int i = t & 15, o = t >> 4;
        float g = wsvb0[o];
        for (int k = 0; k < 100; ++k) g += wsvw0[o * 100 + k] * soL[i][k];
        float sig = 1.0f / (1.0f + expf(-g));
        for (int d = 0; d < 3; ++d) {
            float a = 0.f;
            for (int c = 0; c < 33; ++c) a += wv0[o * 33 + c] * vhL[i][c][d];
            voL[i][o][d] = sig * a;
        }
    }
    __syncthreads();

    // ---- stages c1, c2 : X = [relu(s_prev)(100) | vn(16)] K=116 pad 128
    const float* whx[2]   = { wh1, wh2 };
    const float* bx[2]    = { b1, b2 };
    const float* wsvwx[2] = { wsvw1, wsvw2 };
    const float* wsvbx[2] = { wsvb1, wsvb2 };
    const float* wvx[2]   = { wv1, wv2 };
    const bf16_t* pAx[2]  = { pA1, pA2 };

    for (int st = 0; st < 2; ++st) {
        for (int t = lane; t < 4 * 512; t += 32) Xf[t] = (bf16_t)0.f;
        __syncthreads();
        for (int t = lane; t < 100 * 16; t += 32) {
            int k = t >> 4, i = t & 15;
            Xf[wmma_slot(k, i)] = (bf16_t)fmaxf(soL[i][k], 0.f);
        }
        for (int t = lane; t < 16 * 16; t += 32) {
            int i = t & 15, h = t >> 4;
            float a0 = 0.f, a1 = 0.f, a2 = 0.f;
            for (int c = 0; c < 16; ++c) {
                float w = whx[st][h * 16 + c];
                a0 += w * voL[i][c][0]; a1 += w * voL[i][c][1]; a2 += w * voL[i][c][2];
            }
            vhL[i][h][0] = a0; vhL[i][h][1] = a1; vhL[i][h][2] = a2;
            Xf[wmma_slot(100 + h, i)] = (bf16_t)sqrtf(fmaxf(a0*a0 + a1*a1 + a2*a2, EPSF));
        }
        __syncthreads();

        for (int tile = 0; tile < 7; ++tile) {
            v8f acc = {};
            for (int kc = 0; kc < 4; ++kc) {
                v16bf a = ((const v16bf*)pAx[st])[(tile * 4 + kc) * 32 + lane];
                v16bf bb = *(const v16bf*)&Xf[kc * 512 + lane * 16];
                acc = __builtin_amdgcn_wmma_f32_16x16x32_bf16(
                    false, a, false, bb, (short)0, acc, false, false);
            }
            int i = lane & 15;
            #pragma unroll
            for (int r = 0; r < 8; ++r) {
                int M = tile * 16 + r + 8 * half;
                soL[i][M] = acc[r] + bx[st][M < 100 ? M : 99];
            }
        }
        __syncthreads();

        for (int t = lane; t < 16 * 16; t += 32) {
            int i = t & 15, o = t >> 4;
            float g = wsvbx[st][o];
            for (int k = 0; k < 100; ++k) g += wsvwx[st][o * 100 + k] * soL[i][k];
            float sig = 1.0f / (1.0f + expf(-g));
            for (int d = 0; d < 3; ++d) {
                float a = 0.f;
                for (int c = 0; c < 16; ++c) a += wvx[st][o * 16 + c] * vhL[i][c][d];
                voL[i][o][d] = sig * a;
            }
        }
        __syncthreads();
    }

    // ---- scatter (segment-sum) into dhs/dhv
    for (int t = lane; t < 100 * 16; t += 32) {
        int i = t & 15, c = t >> 4;
        if (base + i < E) atomicAdd(&dhs[(long)dstL[i] * 100 + c], soL[i][c]);
    }
    for (int t = lane; t < 48 * 16; t += 32) {
        int i = t & 15, c = t >> 4;
        if (base + i < E) atomicAdd(&dhv[(long)dstL[i] * 48 + c], voL[i][c / 3][c % 3]);
    }
}

// ---------------------------------------------------------------------------
// Final dense: out = relu(W @ s + b), thread per (node, out-channel)
// ---------------------------------------------------------------------------
__global__ void dense_kernel(const float* sIn, const float* W, const float* b,
                             float* out, int N) {
    long idx = (long)blockIdx.x * blockDim.x + threadIdx.x;
    if (idx >= (long)N * 100) return;
    int i = idx / 100, o = idx % 100;
    float a = b[o];
    const float* sr = sIn + (long)i * 100;
    const float* wr = W + (long)o * 100;
    for (int k = 0; k < 100; ++k) a += wr[k] * sr[k];
    out[idx] = fmaxf(a, 0.f);
}

// ---------------------------------------------------------------------------
// Host-side launch. Input order = jax tree-flatten (sorted keys, recursive):
//  0 edge_index, 1 h_E_s, 2 h_E_v, 3 h_V_s, 4 h_V_v,
//  5 dense_b, 6 dense_w,
//  7 + 34*l : layers[l] -> c0,c1,c2,f0,f1 (each: wh,ws_b,ws_w,wsv_b,wsv_w,wv),
//             n0 (b,w), n1 (b,w)
//  109..111 out_gvp (wh,ws_b,ws_w), 112..113 out_ln (b,w),
//  114..119 we_gvp, 120..121 we_ln (b,w), 122..127 wv_gvp, 128..129 wv_ln (b,w)
// ---------------------------------------------------------------------------
#define LGVP(l, g, f) ((const float*)d_in[7 + 34 * (l) + 6 * (g) + (f)])
#define LLN(l, which, f) ((const float*)d_in[7 + 34 * (l) + 30 + 2 * (which) + (f)])

extern "C" void kernel_launch(void* const* d_in, const int* in_sizes, int n_in,
                              void* d_out, int out_size, void* d_ws, size_t ws_size,
                              hipStream_t stream) {
    const int* eidx     = (const int*)d_in[0];
    const float* hEs    = (const float*)d_in[1];
    const float* hEv    = (const float*)d_in[2];
    const float* hVs    = (const float*)d_in[3];
    const float* hVv    = (const float*)d_in[4];
    const float* denseB = (const float*)d_in[5];
    const float* denseW = (const float*)d_in[6];
    const int E = in_sizes[0] / 2;
    const int N = in_sizes[3] / 6;
    float* outp = (float*)d_out;

    // workspace arena
    char* wsb = (char*)d_ws;
    size_t off = 0;
    auto allocf = [&](size_t n) -> float* {
        off = (off + 255) & ~(size_t)255;
        float* p = (float*)(wsb + off); off += n * sizeof(float); return p;
    };
    auto allocb = [&](size_t n) -> bf16_t* {
        off = (off + 255) & ~(size_t)255;
        bf16_t* p = (bf16_t*)(wsb + off); off += n * sizeof(bf16_t); return p;
    };

    float* t6  = allocf((size_t)N * 6);
    float* t9  = allocf((size_t)N * 9);
    float* sN  = allocf((size_t)N * 100);
    float* vN  = allocf((size_t)N * 48);
    float* te32 = allocf((size_t)E * 32);
    float* te3  = allocf((size_t)E * 3);
    float* esB  = allocf((size_t)E * 32);
    float* evB  = allocf((size_t)E * 3);
    float* cnt  = allocf((size_t)N);
    float* dhs  = allocf((size_t)N * 100);
    float* dhv  = allocf((size_t)N * 48);
    float* fs   = allocf((size_t)N * 400);
    float* fv   = allocf((size_t)N * 96);
    float* gs   = allocf((size_t)N * 100);
    float* gv   = allocf((size_t)N * 48);
    float* osb  = allocf((size_t)N * 100);
    bf16_t* pc0[3]; bf16_t* pc1[3]; bf16_t* pc2[3];
    for (int l = 0; l < 3; ++l) {
        pc0[l] = allocb(7 * 9 * 512);
        pc1[l] = allocb(7 * 4 * 512);
        pc2[l] = allocb(7 * 4 * 512);
    }

    // ---- pack WMMA weights (bf16 A-fragments) for the 9 edge GEMMs
    for (int l = 0; l < 3; ++l) {
        pack_wmma_a_kernel<<<(7*9*512 + 255)/256, 256, 0, stream>>>(
            LGVP(l,0,2), pc0[l], 100, 265, 7, 9);
        pack_wmma_a_kernel<<<(7*4*512 + 255)/256, 256, 0, stream>>>(
            LGVP(l,1,2), pc1[l], 100, 116, 7, 4);
        pack_wmma_a_kernel<<<(7*4*512 + 255)/256, 256, 0, stream>>>(
            LGVP(l,2,2), pc2[l], 100, 116, 7, 4);
    }

    // ---- node embed: LN(6,3) then GVP(6,3 -> 100,16)
    ln_kernel<<<N, 128, 0, stream>>>(hVs, hVv, t6, t9,
        (const float*)d_in[129], (const float*)d_in[128], 6, 3);
    gvp_kernel<<<N, 128, 0, stream>>>(t6, t9, sN, vN,
        (const float*)d_in[122], (const float*)d_in[124], (const float*)d_in[123],
        (const float*)d_in[127], (const float*)d_in[126], (const float*)d_in[125],
        6, 3, 100, 16, 16, 0);

    // ---- edge embed: LN(32,1) then GVP(32,1 -> 32,1)
    ln_kernel<<<E, 128, 0, stream>>>(hEs, hEv, te32, te3,
        (const float*)d_in[121], (const float*)d_in[120], 32, 1);
    gvp_kernel<<<E, 64, 0, stream>>>(te32, te3, esB, evB,
        (const float*)d_in[114], (const float*)d_in[116], (const float*)d_in[115],
        (const float*)d_in[119], (const float*)d_in[118], (const float*)d_in[117],
        32, 1, 32, 1, 1, 0);

    // ---- degree counts
    fzero_kernel<<<(N + 255)/256, 256, 0, stream>>>(cnt, N);
    cnt_scatter_kernel<<<(E + 255)/256, 256, 0, stream>>>(eidx, cnt, E);

    // ---- 3 message-passing layers
    for (int l = 0; l < 3; ++l) {
        fzero_kernel<<<((long)N*100 + 255)/256, 256, 0, stream>>>(dhs, (long)N*100);
        fzero_kernel<<<((long)N*48  + 255)/256, 256, 0, stream>>>(dhv, (long)N*48);

        edge_msg_kernel<<<(E + 15)/16, 32, 0, stream>>>(
            eidx, E, sN, vN, esB, evB,
            pc0[l], pc1[l], pc2[l],
            LGVP(l,0,0), LGVP(l,0,1), LGVP(l,0,4), LGVP(l,0,3), LGVP(l,0,5),
            LGVP(l,1,0), LGVP(l,1,1), LGVP(l,1,4), LGVP(l,1,3), LGVP(l,1,5),
            LGVP(l,2,0), LGVP(l,2,1), LGVP(l,2,4), LGVP(l,2,3), LGVP(l,2,5),
            dhs, dhv);

        // residual + mean aggregation + LN n0
        resadd_kernel<<<((long)N*100 + 255)/256, 256, 0, stream>>>(sN, dhs, cnt, (long)N*100, 100);
        resadd_kernel<<<((long)N*48  + 255)/256, 256, 0, stream>>>(vN, dhv, cnt, (long)N*48, 48);
        ln_kernel<<<N, 128, 0, stream>>>(sN, vN, sN, vN, LLN(l,0,1), LLN(l,0,0), 100, 16);

        // feed-forward f0 (100,16 -> 400,32, relu) ; f1 (400,32 -> 100,16)
        gvp_kernel<<<N, 128, 0, stream>>>(sN, vN, fs, fv,
            LGVP(l,3,0), LGVP(l,3,2), LGVP(l,3,1),
            LGVP(l,3,5), LGVP(l,3,4), LGVP(l,3,3),
            100, 16, 400, 32, 32, 1);
        gvp_kernel<<<N, 128, 0, stream>>>(fs, fv, gs, gv,
            LGVP(l,4,0), LGVP(l,4,2), LGVP(l,4,1),
            LGVP(l,4,5), LGVP(l,4,4), LGVP(l,4,3),
            400, 32, 100, 16, 32, 0);

        resadd_kernel<<<((long)N*100 + 255)/256, 256, 0, stream>>>(sN, gs, nullptr, (long)N*100, 100);
        resadd_kernel<<<((long)N*48  + 255)/256, 256, 0, stream>>>(vN, gv, nullptr, (long)N*48, 48);
        ln_kernel<<<N, 128, 0, stream>>>(sN, vN, sN, vN, LLN(l,1,1), LLN(l,1,0), 100, 16);
    }

    // ---- output head: LN, GVP(100,16 -> 100, vo=0, relu), dense+relu
    ln_kernel<<<N, 128, 0, stream>>>(sN, vN, sN, vN,
        (const float*)d_in[113], (const float*)d_in[112], 100, 16);
    gvp_kernel<<<N, 128, 0, stream>>>(sN, vN, osb, nullptr,
        (const float*)d_in[109], (const float*)d_in[111], (const float*)d_in[110],
        nullptr, nullptr, nullptr,
        100, 16, 100, 0, 16, 1);
    dense_kernel<<<((long)N*100 + 255)/256, 256, 0, stream>>>(osb, denseW, denseB, outp, N);
}
